// SSD300_7112465842583
// MI455X (gfx1250) — compile-verified, matrix operand order checked by reading, not run on hardware
//
#include <hip/hip_runtime.h>
#include <math.h>

// ---------------------------------------------------------------------------
// SSD300 forward on gfx1250 (MI455X), f16-NHWC dataflow.
// Convs = implicit GEMM on v_wmma_f32_16x16x32_f16 (wave32, fp32 accum).
// - Activations: f16 NHWC (channel-fastest) -> each lane's WMMA B operand is
//   two contiguous 16-byte runs (global_load_b128 x2, zero converts).
// - Weights: repacked per layer to f16 [kyx][Mpad][C] -> LDS A-stage is pure
//   b128 copies; A operand reads are ds_load_b128 x2.
// - C/D fragment: 8 consecutive M rows per lane -> one b128 NHWC store.
// GEMM-K ordered (ky,kx,c); C%64==0 for all non-stem layers.
// ---------------------------------------------------------------------------

typedef __attribute__((ext_vector_type(16))) _Float16 v16h;
typedef __attribute__((ext_vector_type(8)))  _Float16 v8h;
typedef __attribute__((ext_vector_type(8)))  float    v8f;

#define CONV_BLOCK 128
#define MTILES 4            // 4 x 16 = 64 output channels per block
#define NSUB 2              // 2 x 16 pixels per wave -> 128 pixels per block
#define ROWP 72             // LDS row pitch (f16) for bank spread

union Vh { v16h v; v8h h8[2]; unsigned int u[8]; _Float16 h[16]; };

// ---------------------------------------------------------------------------
// Weight repack: fp32 OIHW -> f16 [kyx][Mpad][C], zero-padded m >= O
// ---------------------------------------------------------------------------
__global__ void repack_w(const float* __restrict__ w, _Float16* __restrict__ wp,
                         int O, int C, int KK, int Mpad)
{
  const long long total = (long long)KK * Mpad * C;
  const long long i = (long long)blockIdx.x * blockDim.x + threadIdx.x;
  if (i >= total) return;
  const int c = (int)(i % C);
  long long t = i / C;
  const int m = (int)(t % Mpad);
  const int kyx = (int)(t / Mpad);
  float v = 0.f;
  if (m < O) v = w[((size_t)m * C + c) * KK + kyx];
  wp[i] = (_Float16)v;
}

// ---------------------------------------------------------------------------
// Main conv kernel (C % 64 == 0). Block = 4 waves, tile 64(M) x 128(N).
// in: f16 NHWC; wp: f16 [kyx][Mpad][C]; out: f16 NHWC with Mpad channels.
// ---------------------------------------------------------------------------
__global__ __launch_bounds__(CONV_BLOCK)
void conv_wmma_nhwc(const _Float16* __restrict__ in, const _Float16* __restrict__ wp,
                    const float* __restrict__ bias, _Float16* __restrict__ out,
                    int Bn, int C, int H, int W, int O, int Ho, int Wo,
                    int Kk, int stride, int pad, int dil, int doRelu, int Mpad)
{
  __shared__ __align__(16) _Float16 As[64 * ROWP];

  const int tid  = threadIdx.x;
  const int lane = tid & 31;
  const int wave = tid >> 5;
  const int tileM = blockIdx.x;               // 64 output channels
  const int nbase = (blockIdx.y * 4 + wave) * (16 * NSUB);
  const int HoWo = Ho * Wo;
  const int Npix = Bn * HoWo;
  const int KK = Kk * Kk;
  const size_t wplane = (size_t)Mpad * C;

  const int mn   = lane & 15;
  const int hsel = (lane >> 4) & 1;
  const int hs8  = hsel * 8;

  // this lane's NSUB output pixels
  bool vn[NSUB]; int ng[NSUB], pb[NSUB], poy[NSUB], pox[NSUB];
#pragma unroll
  for (int j = 0; j < NSUB; ++j) {
    ng[j] = nbase + j * 16 + mn;
    vn[j] = (ng[j] < Npix);
    const int ngc = vn[j] ? ng[j] : 0;
    pb[j]  = ngc / HoWo;
    const int r1 = ngc - pb[j] * HoWo;
    poy[j] = r1 / Wo;
    pox[j] = r1 - poy[j] * Wo;
  }

  v8f acc[MTILES][NSUB];
#pragma unroll
  for (int t = 0; t < MTILES; ++t)
#pragma unroll
    for (int j = 0; j < NSUB; ++j)
      acc[t][j] = (v8f){0.f, 0.f, 0.f, 0.f, 0.f, 0.f, 0.f, 0.f};

  for (int ky = 0; ky < Kk; ++ky) {
    for (int kx = 0; kx < Kk; ++kx) {
      const int kyx = ky * Kk + kx;
      const _Float16* wbase = wp + (size_t)kyx * wplane + (size_t)(tileM * 64) * C;

      // per-pixel spatial tap state
      bool inb[NSUB]; const _Float16* pix[NSUB];
#pragma unroll
      for (int j = 0; j < NSUB; ++j) {
        const int iy = poy[j] * stride - pad + ky * dil;
        const int ix = pox[j] * stride - pad + kx * dil;
        inb[j] = vn[j] && (iy >= 0) && (iy < H) && (ix >= 0) && (ix < W);
        pix[j] = in + ((size_t)(pb[j] * H + iy) * W + ix) * C;  // deref only if inb
      }

      for (int cb = 0; cb < C; cb += 64) {
        if (cb + 64 < C)
          __builtin_prefetch(wbase + cb + 64, 0, 1);

        // ---- A stage: 64 m x 64 c f16, pure b128 copies ------------------
#pragma unroll
        for (int i = 0; i < 4; ++i) {
          const int idx8 = tid * 4 + i;          // 0..511
          const int ml  = idx8 >> 3;             // 0..63
          const int cl  = (idx8 & 7) * 8;        // 0..56
          *(v8h*)&As[ml * ROWP + cl] = *(const v8h*)(wbase + (size_t)ml * C + cb + cl);
        }
        __syncthreads();

        // ---- two 32-wide K sub-steps ------------------------------------
#pragma unroll
        for (int s2 = 0; s2 < 2; ++s2) {
          const int co = cb + 32 * s2;
          Vh Bop[NSUB];
#pragma unroll
          for (int j = 0; j < NSUB; ++j) {
            if (inb[j]) {
              const _Float16* pj = pix[j] + co;
              Bop[j].h8[0] = *(const v8h*)(pj + hs8);
              Bop[j].h8[1] = *(const v8h*)(pj + 16 + hs8);
            } else {
#pragma unroll
              for (int q = 0; q < 8; ++q) Bop[j].u[q] = 0;
            }
          }
#pragma unroll
          for (int t = 0; t < MTILES; ++t) {
            Vh Aop;
            const int row = (t * 16 + mn) * ROWP + 32 * s2;
            Aop.h8[0] = *(const v8h*)&As[row + hs8];
            Aop.h8[1] = *(const v8h*)&As[row + 16 + hs8];
#pragma unroll
            for (int j = 0; j < NSUB; ++j)
              acc[t][j] = __builtin_amdgcn_wmma_f32_16x16x32_f16(
                  false, Aop.v, false, Bop[j].v, (short)0, acc[t][j], false, false);
          }
        }
        __syncthreads();
      }
    }
  }

  // ---- epilogue: bias (+relu), one b128 NHWC store per (tile, pixel) -----
#pragma unroll
  for (int t = 0; t < MTILES; ++t) {
    const int mbase = tileM * 64 + t * 16 + hs8;
    float bs[8];
#pragma unroll
    for (int r = 0; r < 8; ++r) {
      const int m = mbase + r;
      bs[r] = (m < O) ? bias[m] : 0.f;
    }
#pragma unroll
    for (int j = 0; j < NSUB; ++j) {
      if (vn[j]) {
        v8h hv;
#pragma unroll
        for (int r = 0; r < 8; ++r) {
          float vv = acc[t][j][r] + bs[r];
          if (doRelu) vv = vv > 0.f ? vv : 0.f;
          hv[r] = (_Float16)vv;
        }
        *(v8h*)(out + (size_t)ng[j] * Mpad + mbase) = hv;
      }
    }
  }
}

// ---------------------------------------------------------------------------
// Stem conv: fp32 NCHW input (C=3, 3x3, pad 1), f16 NHWC output (64 ch).
// Kdim=27 -> single WMMA per 16x16 tile; compile-time constant k-mapping.
// ---------------------------------------------------------------------------
__global__ __launch_bounds__(CONV_BLOCK)
void conv_first_nhwc(const float* __restrict__ in, const float* __restrict__ wgt,
                     const float* __restrict__ bias, _Float16* __restrict__ out,
                     int Bn, int H, int W, int Ho, int Wo)
{
  const int lane = threadIdx.x & 31;
  const int wave = threadIdx.x >> 5;
  const int tileM = blockIdx.x;                // 16 of 64 output channels
  const int tileN = blockIdx.y * 4 + wave;
  const int HoWo = Ho * Wo;
  const int Npix = Bn * HoWo;
  const int mn   = lane & 15;
  const int hsel = (lane >> 4) & 1;

  const int ng  = tileN * 16 + mn;
  const bool vn = (ng < Npix);
  const int ngc = vn ? ng : 0;
  const int b   = ngc / HoWo;
  const int r1  = ngc - b * HoWo;
  const int oy  = r1 / Wo;
  const int ox  = r1 - oy * Wo;
  const int iy0 = oy - 1, ix0 = ox - 1;
  const int mgA = tileM * 16 + mn;

  Vh Aop, Bop;
#pragma unroll
  for (int v = 0; v < 8; ++v) {
#pragma unroll
    for (int t = 0; t < 2; ++t) {
      const int k = ((v & 3) * 2 + hsel * 8 + ((v >> 2) * 16)) + t;
      float av = 0.f, bv = 0.f;
      if (k < 27) {
        av = wgt[mgA * 27 + k];
        if (vn) {
          const int c  = k / 9;
          const int r2 = k - c * 9;
          const int ky = r2 / 3;
          const int kx = r2 - ky * 3;
          const int iy = iy0 + ky, ix = ix0 + kx;
          if (iy >= 0 && iy < H && ix >= 0 && ix < W)
            bv = in[((size_t)(b * 3 + c) * H + iy) * W + ix];
        }
      }
      Aop.h[2 * v + t] = (_Float16)av;
      Bop.h[2 * v + t] = (_Float16)bv;
    }
  }
  v8f acc = {0.f, 0.f, 0.f, 0.f, 0.f, 0.f, 0.f, 0.f};
  acc = __builtin_amdgcn_wmma_f32_16x16x32_f16(false, Aop.v, false, Bop.v,
                                               (short)0, acc, false, false);
  if (vn) {
    const int mbase = tileM * 16 + hsel * 8;
    v8h hv;
#pragma unroll
    for (int r = 0; r < 8; ++r) {
      float vv = acc[r] + bias[mbase + r];
      vv = vv > 0.f ? vv : 0.f;
      hv[r] = (_Float16)vv;
    }
    *(v8h*)(out + (size_t)ng * 64 + mbase) = hv;
  }
}

// ---------------------------------------------------------------------------
// Max-pool, NHWC f16 (channel-fastest -> coalesced). OOB window cells skip.
// ---------------------------------------------------------------------------
__global__ void maxpool_nhwc(const _Float16* __restrict__ in, _Float16* __restrict__ out,
                             int Bn, int C, int H, int W, int Ho, int Wo,
                             int k, int s, int pad)
{
  const long long total = (long long)Bn * Ho * Wo * C;
  const long long i = (long long)blockIdx.x * blockDim.x + threadIdx.x;
  if (i >= total) return;
  const int c = (int)(i % C); long long t = i / C;
  const int ox = (int)(t % Wo); t /= Wo;
  const int oy = (int)(t % Ho);
  const int b  = (int)(t / Ho);
  float m = -3.402823466e38f;
  const int y0 = oy * s - pad, x0 = ox * s - pad;
  for (int dy = 0; dy < k; ++dy) {
    const int iy = y0 + dy; if (iy < 0 || iy >= H) continue;
    for (int dx = 0; dx < k; ++dx) {
      const int ix = x0 + dx; if (ix < 0 || ix >= W) continue;
      const float v = (float)in[((size_t)(b * H + iy) * W + ix) * C + c];
      m = v > m ? v : m;
    }
  }
  out[i] = (_Float16)m;
}

// ---------------------------------------------------------------------------
// SSD L2Norm, NHWC f16: channels contiguous per pixel.
// ---------------------------------------------------------------------------
__global__ void l2norm_nhwc(const _Float16* __restrict__ in, const float* __restrict__ w,
                            _Float16* __restrict__ out, int NP, int C)
{
  const int i = blockIdx.x * blockDim.x + threadIdx.x;
  if (i >= NP) return;
  const _Float16* src = in + (size_t)i * C;
  float s = 0.f;
  for (int c = 0; c < C; ++c) { const float v = (float)src[c]; s += v * v; }
  const float inv = 1.0f / (sqrtf(s) + 1e-10f);
  _Float16* dst = out + (size_t)i * C;
  for (int c = 0; c < C; ++c) dst[c] = (_Float16)((float)src[c] * inv * w[c]);
}

// ---------------------------------------------------------------------------
// Head scatter: padded NHWC f16 conv result -> fp32 slice of d_out
// (NHWC is exactly the reference's transpose(0,2,3,1) flattening).
// ---------------------------------------------------------------------------
__global__ void scatter_head(const _Float16* __restrict__ in, float* __restrict__ out,
                             int Bn, int Cb, int FF, int Mpad, int batch_stride, int off)
{
  const int total = Bn * FF * Cb;
  const int i = blockIdx.x * blockDim.x + threadIdx.x;
  if (i >= total) return;
  const int c  = i % Cb;
  const int t  = i / Cb;
  const int yx = t % FF;
  const int b  = t / FF;
  out[(size_t)b * batch_stride + off + (size_t)yx * Cb + c] =
      (float)in[((size_t)b * FF + yx) * Mpad + c];
}

__global__ void copy_k(const float* __restrict__ in, float* __restrict__ out, int n)
{
  const int i = blockIdx.x * blockDim.x + threadIdx.x;
  if (i < n) out[i] = in[i];
}

// ---------------------------------------------------------------------------
// Host driver
// ---------------------------------------------------------------------------
extern "C" void kernel_launch(void* const* d_in, const int* in_sizes, int n_in,
                              void* d_out, int out_size, void* d_ws, size_t ws_size,
                              hipStream_t stream) {
  (void)in_sizes; (void)n_in; (void)out_size; (void)ws_size;
  const int Bn = 8;

  // ---- inputs (setup_inputs order, lists flattened) ----
  const float* x = (const float*)d_in[0];
  const float* vw[15]; const float* vb[15];
  for (int i = 0; i < 15; ++i) { vw[i] = (const float*)d_in[1 + i]; vb[i] = (const float*)d_in[16 + i]; }
  const float* l2w = (const float*)d_in[31];
  const float* ew[8]; const float* eb[8];
  for (int i = 0; i < 8; ++i) { ew[i] = (const float*)d_in[32 + i]; eb[i] = (const float*)d_in[40 + i]; }
  const float* lw[6]; const float* lb[6]; const float* cw[6]; const float* cb[6];
  for (int i = 0; i < 6; ++i) {
    lw[i] = (const float*)d_in[48 + i]; lb[i] = (const float*)d_in[54 + i];
    cw[i] = (const float*)d_in[60 + i]; cb[i] = (const float*)d_in[66 + i];
  }
  const float* priors = (const float*)d_in[72];

  // ---- workspace layout (f16, 256B-aligned chunks) ----
  unsigned char* wsb = (unsigned char*)d_ws;
  size_t off = 0;
  auto allocH = [&](size_t nh) {
    _Float16* p = (_Float16*)(wsb + off);
    off += ((nh * sizeof(_Float16) + 255) / 256) * 256;
    return p;
  };
  _Float16* A  = allocH((size_t)Bn * 300 * 300 * 64);   // ping
  _Float16* Bb = allocH((size_t)Bn * 300 * 300 * 64);   // pong
  _Float16* S0 = allocH((size_t)Bn * 38 * 38 * 512);
  _Float16* S1 = allocH((size_t)Bn * 19 * 19 * 1024);
  _Float16* S2 = allocH((size_t)Bn * 10 * 10 * 512);
  _Float16* S3 = allocH((size_t)Bn * 5 * 5 * 256);
  _Float16* S4 = allocH((size_t)Bn * 3 * 3 * 256);
  _Float16* S5 = allocH((size_t)Bn * 1 * 1 * 256);
  _Float16* HT = allocH((size_t)Bn * 38 * 38 * 128);    // largest padded head out
  _Float16* WP = allocH((size_t)5000000);               // repacked weights (max 4.72M)

  // C % 64 == 0 for every call below
  auto conv = [&](const _Float16* pin, const float* pw, const float* pb, _Float16* pout,
                  int C, int H, int W, int O, int Ho, int Wo,
                  int K, int s, int p, int d, int relu) {
    const int KK = K * K;
    const int Mpad = ((O + 63) / 64) * 64;
    const long long wtot = (long long)KK * Mpad * C;
    repack_w<<<dim3((unsigned)((wtot + 255) / 256)), 256, 0, stream>>>(
        pw, WP, O, C, KK, Mpad);
    const int Npix = Bn * Ho * Wo;
    dim3 grid((unsigned)(Mpad / 64), (unsigned)((Npix + 127) / 128));
    conv_wmma_nhwc<<<grid, CONV_BLOCK, 0, stream>>>(pin, WP, pb, pout,
        Bn, C, H, W, O, Ho, Wo, K, s, p, d, relu, Mpad);
  };
  auto pool = [&](const _Float16* pin, _Float16* pout, int C, int H, int W,
                  int Ho, int Wo, int k, int s, int p) {
    const long long total = (long long)Bn * Ho * Wo * C;
    maxpool_nhwc<<<dim3((unsigned)((total + 255) / 256)), 256, 0, stream>>>(
        pin, pout, Bn, C, H, W, Ho, Wo, k, s, p);
  };

  // ---- VGG trunk ----
  {
    const int Npix = Bn * 300 * 300;
    dim3 grid(4u, (unsigned)((Npix + 63) / 64));
    conv_first_nhwc<<<grid, CONV_BLOCK, 0, stream>>>(x, vw[0], vb[0], A,
        Bn, 300, 300, 300, 300);
  }
  conv(A,  vw[1], vb[1], Bb, 64,  300, 300, 64,  300, 300, 3, 1, 1, 1, 1);
  pool(Bb, A, 64, 300, 300, 150, 150, 2, 2, 0);
  conv(A,  vw[2], vb[2], Bb, 64,  150, 150, 128, 150, 150, 3, 1, 1, 1, 1);
  conv(Bb, vw[3], vb[3], A,  128, 150, 150, 128, 150, 150, 3, 1, 1, 1, 1);
  pool(A, Bb, 128, 150, 150, 75, 75, 2, 2, 0);
  conv(Bb, vw[4], vb[4], A,  128, 75, 75, 256, 75, 75, 3, 1, 1, 1, 1);
  conv(A,  vw[5], vb[5], Bb, 256, 75, 75, 256, 75, 75, 3, 1, 1, 1, 1);
  conv(Bb, vw[6], vb[6], A,  256, 75, 75, 256, 75, 75, 3, 1, 1, 1, 1);
  pool(A, Bb, 256, 75, 75, 38, 38, 2, 2, 0);        // ceil-mode: Ho=38, clamp
  conv(Bb, vw[7], vb[7], A,  256, 38, 38, 512, 38, 38, 3, 1, 1, 1, 1);
  conv(A,  vw[8], vb[8], Bb, 512, 38, 38, 512, 38, 38, 3, 1, 1, 1, 1);
  conv(Bb, vw[9], vb[9], A,  512, 38, 38, 512, 38, 38, 3, 1, 1, 1, 1); // conv4_3
  {
    const int np = Bn * 38 * 38;
    l2norm_nhwc<<<dim3((np + 255) / 256), 256, 0, stream>>>(A, l2w, S0, np, 512);
  }
  pool(A, Bb, 512, 38, 38, 19, 19, 2, 2, 0);
  conv(Bb, vw[10], vb[10], A,  512, 19, 19, 512, 19, 19, 3, 1, 1, 1, 1);
  conv(A,  vw[11], vb[11], Bb, 512, 19, 19, 512, 19, 19, 3, 1, 1, 1, 1);
  conv(Bb, vw[12], vb[12], A,  512, 19, 19, 512, 19, 19, 3, 1, 1, 1, 1);
  pool(A, Bb, 512, 19, 19, 19, 19, 3, 1, 1);
  conv(Bb, vw[13], vb[13], A,  512, 19, 19, 1024, 19, 19, 3, 1, 6, 6, 1); // dil-6
  conv(A,  vw[14], vb[14], S1, 1024, 19, 19, 1024, 19, 19, 1, 1, 0, 1, 1); // conv7

  // ---- extras ----
  conv(S1, ew[0], eb[0], A,  1024, 19, 19, 256, 19, 19, 1, 1, 0, 1, 1);
  conv(A,  ew[1], eb[1], S2, 256,  19, 19, 512, 10, 10, 3, 2, 1, 1, 1);
  conv(S2, ew[2], eb[2], A,  512,  10, 10, 128, 10, 10, 1, 1, 0, 1, 1);
  conv(A,  ew[3], eb[3], S3, 128,  10, 10, 256, 5,  5,  3, 2, 1, 1, 1);
  conv(S3, ew[4], eb[4], A,  256,  5,  5,  128, 5,  5,  1, 1, 0, 1, 1);
  conv(A,  ew[5], eb[5], S4, 128,  5,  5,  256, 3,  3,  3, 1, 0, 1, 1);
  conv(S4, ew[6], eb[6], A,  256,  3,  3,  128, 3,  3,  1, 1, 0, 1, 1);
  conv(A,  ew[7], eb[7], S5, 128,  3,  3,  256, 1,  1,  3, 1, 0, 1, 1);

  // ---- heads ----
  const _Float16* srcs[6] = {S0, S1, S2, S3, S4, S5};
  const int fm[6]        = {38, 19, 10, 5, 3, 1};
  const int chn[6]       = {512, 1024, 512, 256, 256, 256};
  const int bx[6]        = {4, 6, 6, 6, 4, 4};
  const int locOff[6]    = {0, 23104, 31768, 34168, 34768, 34912};
  const int confOff[6]   = {0, 121296, 166782, 179382, 182532, 183288};
  const int LOC_STRIDE   = 34928;   // 8732*4 per batch
  const int CONF_STRIDE  = 183372;  // 8732*21 per batch

  float* outF     = (float*)d_out;
  float* locBase  = outF;
  float* confBase = outF + (size_t)Bn * LOC_STRIDE;
  float* priBase  = confBase + (size_t)Bn * CONF_STRIDE;

  for (int s = 0; s < 6; ++s) {
    const int F = fm[s], FF = F * F;
    const int Cl = bx[s] * 4;
    const int Cc = bx[s] * 21;
    const int MpL = ((Cl + 63) / 64) * 64;
    const int MpC = ((Cc + 63) / 64) * 64;
    conv(srcs[s], lw[s], lb[s], HT, chn[s], F, F, Cl, F, F, 3, 1, 1, 1, 0);
    {
      const int n = Bn * FF * Cl;
      scatter_head<<<dim3((n + 255) / 256), 256, 0, stream>>>(
          HT, locBase, Bn, Cl, FF, MpL, LOC_STRIDE, locOff[s]);
    }
    conv(srcs[s], cw[s], cb[s], HT, chn[s], F, F, Cc, F, F, 3, 1, 1, 1, 0);
    {
      const int n = Bn * FF * Cc;
      scatter_head<<<dim3((n + 255) / 256), 256, 0, stream>>>(
          HT, confBase, Bn, Cc, FF, MpC, CONF_STRIDE, confOff[s]);
    }
  }

  // ---- priors passthrough ----
  copy_k<<<dim3((34928 + 255) / 256), 256, 0, stream>>>(priors, priBase, 34928);
}
